// SpatialAttention_76570676953612
// MI455X (gfx1250) — matrix operand order repaired and checked by reading, not compiled, not run on hardware
//
#include <hip/hip_runtime.h>
#include <hip/hip_bf16.h>

// ---------------------------------------------------------------------------
// Problem constants (from reference)
// ---------------------------------------------------------------------------
constexpr int Bn  = 16;
constexpr int Tn  = 1024;
constexpr int Dn  = 1024;
constexpr int Hn  = 16;
constexpr int Gn  = 4;
constexpr int HDn = 64;                       // head dim
constexpr int QKV_LD = Hn*HDn + 2*Gn*HDn;     // 1536
constexpr float SCALE = 0.125f;
constexpr float THETA = 10000.0f;
constexpr float LOG2E = 1.44269504088896340736f;

typedef __attribute__((ext_vector_type(16))) __bf16 v16bf;
typedef __attribute__((ext_vector_type(8)))  float  v8f;
typedef __attribute__((ext_vector_type(4)))  unsigned int u32x4;
typedef __attribute__((ext_vector_type(8)))  int  i32x8;
typedef __attribute__((ext_vector_type(4)))  int  i32x4;

#if __has_builtin(__builtin_amdgcn_tensor_load_to_lds) && \
    __has_builtin(__builtin_amdgcn_s_wait_tensorcnt)
#define HAVE_TDM 1
#else
#define HAVE_TDM 0
#endif

// A-fragment element->K map for 16-bit 16x32 A tiles (ISA 7.12.2):
// lanes 0-15 hold K {0..7,16..23}, lanes 16-31 hold K {8..15,24..31}
__device__ __forceinline__ int amap(int i, int hi) {
    return (i < 8) ? (hi * 8 + i) : (16 + hi * 8 + (i - 8));
}

__device__ __forceinline__ float hmax16(float v) {
    #pragma unroll
    for (int off = 1; off < 16; off <<= 1)
        v = fmaxf(v, __shfl_xor(v, off, 32));
    return v;
}

#if HAVE_TDM
// ---------------------------------------------------------------------------
// TDM 2D tile load: global (rows x w elems of bf16, row stride = stride elems)
// -> LDS at byte offset lds_off, written row-major with pad bytes appended per
// row so it matches a [rows][w + pad_elems] LDS array. padw0 packs
// pad_enable/pad_interval/pad_amount into D# group1 dword0 (ISA ch. 8.4).
// 6-arg builtin form (clang-23 / therock-10.0 headers): groups 2/3 + iter
// group passed as zero vectors (2D tensor, no iteration).
// ---------------------------------------------------------------------------
__device__ __forceinline__ void tdm_load_2d(const void* g, unsigned lds_off,
                                            unsigned w, unsigned rows,
                                            unsigned stride, unsigned padw0) {
    unsigned long long ga = (unsigned long long)(uintptr_t)g;
    u32x4 g0;
    g0[0] = 1u;                                   // count=1, user descriptor
    g0[1] = lds_off;                              // LDS byte address
    g0[2] = (unsigned)ga;                         // global_addr[31:0]
    g0[3] = ((unsigned)(ga >> 32) & 0x1ffffffu)   // global_addr[56:32]
          | (2u << 30);                           // type = 2 ("image")
    i32x8 g1;
    g1[0] = (int)(0x00010000u | padw0);           // data_size=1 (2B) | pad cfg
    g1[1] = (int)(w << 16);                       // tensor_dim0[15:0]
    g1[2] = (int)((w >> 16) | (rows << 16));      // tensor_dim0[31:16] | tensor_dim1[15:0]
    g1[3] = (int)((rows >> 16) | (w << 16));      // tensor_dim1[31:16] | tile_dim0
    g1[4] = (int)rows;                            // tile_dim1 | tile_dim2=0
    g1[5] = (int)stride;                          // tensor_dim0_stride[31:0]
    g1[6] = 0;                                    // stride[47:32] | dim1_stride lo
    g1[7] = 0;
    i32x4 z4 = {0, 0, 0, 0};
    i32x8 z8 = {0, 0, 0, 0, 0, 0, 0, 0};
    __builtin_amdgcn_tensor_load_to_lds(g0, g1, z4, z4, z8, 0);
}
// pad cfg: rows of 64 bf16 (32 dwords) + 16B pad: interval 2^(4+1)=32dw, amount 4dw
constexpr unsigned PAD64 = (1u << 20) | (4u << 22) | (3u << 25);
// pad cfg: rows of 32 bf16 (16 dwords) + 16B pad: interval 2^(3+1)=16dw, amount 4dw
constexpr unsigned PAD32 = (1u << 20) | (3u << 22) | (3u << 25);
#endif

// ---------------------------------------------------------------------------
// f32 -> bf16 convert
// ---------------------------------------------------------------------------
__global__ void f32_to_bf16_kernel(const float* __restrict__ in,
                                   __bf16* __restrict__ out, int n) {
    int i = blockIdx.x * blockDim.x + threadIdx.x;
    int stride = gridDim.x * blockDim.x;
    for (; i < n; i += stride) out[i] = (__bf16)in[i];
}

// ---------------------------------------------------------------------------
// Tiled bf16 WMMA GEMM:  C[M,N] = A[M,K] @ B[K,N]   (row-major, bf16 in)
// 128 threads = 4 waves; block tile 64x64; each wave owns a 32x32 quadrant.
// K stepped by 32; tiles staged in LDS, double-buffered via TDM when present.
// ---------------------------------------------------------------------------
template <typename OutT>
__global__ __launch_bounds__(128)
void gemm_bf16_wmma(const __bf16* __restrict__ A, const __bf16* __restrict__ Bm,
                    OutT* __restrict__ C, int M, int N, int K) {
    __shared__ __align__(16) __bf16 As[2][64][40];   // 32 cols + 8 pad
    __shared__ __align__(16) __bf16 Bs[2][32][72];   // 64 cols + 8 pad

    const int tid  = threadIdx.x;
    const int lane = tid & 31;
    const int wave = tid >> 5;
    const int l  = lane & 15;
    const int hi = lane >> 4;
    const int wr = (wave >> 1) * 32;
    const int wc = (wave & 1)  * 32;
    const int row0 = blockIdx.y * 64;
    const int col0 = blockIdx.x * 64;
    const int NK = K / 32;

    v8f acc[2][2] = {};

#if HAVE_TDM
    if (wave == 0) {
        tdm_load_2d(A + (size_t)row0 * K, (unsigned)(uintptr_t)&As[0][0][0],
                    32, 64, (unsigned)K, PAD32);
        tdm_load_2d(Bm + col0, (unsigned)(uintptr_t)&Bs[0][0][0],
                    64, 32, (unsigned)N, PAD64);
    }
#endif

    for (int i = 0; i < NK; ++i) {
        const int cur = i & 1;
#if HAVE_TDM
        if (wave == 0) {
            if (i + 1 < NK) {
                const int nxt = cur ^ 1;
                tdm_load_2d(A + (size_t)row0 * K + (i + 1) * 32,
                            (unsigned)(uintptr_t)&As[nxt][0][0],
                            32, 64, (unsigned)K, PAD32);
                tdm_load_2d(Bm + (size_t)((i + 1) * 32) * N + col0,
                            (unsigned)(uintptr_t)&Bs[nxt][0][0],
                            64, 32, (unsigned)N, PAD64);
                __builtin_amdgcn_s_wait_tensorcnt(2);  // tiles for step i landed
            } else {
                __builtin_amdgcn_s_wait_tensorcnt(0);
            }
        }
        __syncthreads();
#else
        #pragma unroll
        for (int it = 0; it < 2; ++it) {
            int idx = tid + it * 128;
            int r = idx >> 2, c = idx & 3;
            *reinterpret_cast<uint4*>(&As[cur][r][c * 8]) =
                *reinterpret_cast<const uint4*>(A + (size_t)(row0 + r) * K + i * 32 + c * 8);
        }
        #pragma unroll
        for (int it = 0; it < 2; ++it) {
            int idx = tid + it * 128;
            int r = idx >> 3, c = idx & 7;
            *reinterpret_cast<uint4*>(&Bs[cur][r][c * 8]) =
                *reinterpret_cast<const uint4*>(Bm + (size_t)(i * 32 + r) * N + col0 + c * 8);
        }
        __syncthreads();
#endif

        v16bf fa[2], fb[2];
        #pragma unroll
        for (int mi = 0; mi < 2; ++mi)
            #pragma unroll
            for (int ii = 0; ii < 16; ++ii)
                fa[mi][ii] = As[cur][wr + mi * 16 + l][amap(ii, hi)];
        #pragma unroll
        for (int ni = 0; ni < 2; ++ni)
            #pragma unroll
            for (int ii = 0; ii < 16; ++ii)
                fb[ni][ii] = Bs[cur][amap(ii, hi)][wc + ni * 16 + l];

        #pragma unroll
        for (int mi = 0; mi < 2; ++mi)
            #pragma unroll
            for (int ni = 0; ni < 2; ++ni)
                acc[mi][ni] = __builtin_amdgcn_wmma_f32_16x16x32_bf16(
                    false, fa[mi], false, fb[ni], (short)0, acc[mi][ni], false, false);
        __syncthreads();
    }

    #pragma unroll
    for (int mi = 0; mi < 2; ++mi)
        #pragma unroll
        for (int ni = 0; ni < 2; ++ni)
            #pragma unroll
            for (int r = 0; r < 8; ++r)
                C[(size_t)(row0 + wr + mi * 16 + r + hi * 8) * N
                  + col0 + wc + ni * 16 + l] = (OutT)acc[mi][ni][r];
}

// ---------------------------------------------------------------------------
// 2D RoPE in place on q (cols 0..1023) and k (cols 1024..1279) of qkv (bf16).
// ---------------------------------------------------------------------------
__global__ __launch_bounds__(320)
void rope2d_kernel(__bf16* __restrict__ qkv) {
    const int row  = blockIdx.x;           // b*T + t
    const int u    = threadIdx.x;          // 0..319
    const int head = u >> 4;               // 0..15 q heads, 16..19 k groups
    const int j    = u & 15;               // freq index
    const int t    = row & (Tn - 1);
    const float gx = (float)(t >> 5);
    const float gy = (float)(t & 31);

    const float inv = __powf(THETA, -(float)j / 16.0f);
    float sx, cx, sy, cy;
    __sincosf(gx * inv, &sx, &cx);
    __sincosf(gy * inv, &sy, &cy);

    const int cb = (head < Hn) ? head * HDn : (Hn * HDn + (head - Hn) * HDn);
    __bf16* p = qkv + (size_t)row * QKV_LD + cb;

    float x1 = (float)p[j],      x2 = (float)p[16 + j];
    float y1 = (float)p[32 + j], y2 = (float)p[48 + j];
    p[j]      = (__bf16)(x1 * cx - x2 * sx);
    p[16 + j] = (__bf16)(x1 * sx + x2 * cx);
    p[32 + j] = (__bf16)(y1 * cy - y2 * sy);
    p[48 + j] = (__bf16)(y1 * sy + y2 * cy);
}

// ---------------------------------------------------------------------------
// Flash attention: one wave per (b, h, 16-row q tile). K/V tiles (32x64 bf16)
// double-buffered through LDS by the TDM. Softmax denominator computed by an
// extra WMMA against a constant ones-column B fragment (P @ 1), so only the
// row-max needs cross-lane shuffles.
// ---------------------------------------------------------------------------
__global__ __launch_bounds__(32)
void attn_kernel(const __bf16* __restrict__ qkv, __bf16* __restrict__ ao) {
    __shared__ __align__(16) __bf16 Qs[16][72];
    __shared__ __align__(16) __bf16 Ks[2][32][72];
    __shared__ __align__(16) __bf16 Vs[2][32][72];
    __shared__ __align__(16) __bf16 Ps[16][40];

    const int qt = blockIdx.x;             // 0..63
    const int h  = blockIdx.y;             // 0..15
    const int b  = blockIdx.z;             // 0..15
    const int g  = h >> 2;                 // kv group (GQA repeat)
    const int lane = threadIdx.x;
    const int l  = lane & 15;
    const int hi = lane >> 4;
    const int t0 = qt * 16;
    const size_t rowbase = (size_t)(b * Tn) * QKV_LD;
    const __bf16* kbase = qkv + rowbase + Hn * HDn + g * HDn;
    const __bf16* vbase = qkv + rowbase + Hn * HDn + Gn * HDn + g * HDn;
    const int NT = Tn / 32;

#if HAVE_TDM
    // Q tile + first K/V tiles via TDM (in-order completion on TENSORcnt)
    tdm_load_2d(qkv + rowbase + (size_t)t0 * QKV_LD + h * HDn,
                (unsigned)(uintptr_t)&Qs[0][0], 64, 16, QKV_LD, PAD64);
    tdm_load_2d(kbase, (unsigned)(uintptr_t)&Ks[0][0][0], 64, 32, QKV_LD, PAD64);
    tdm_load_2d(vbase, (unsigned)(uintptr_t)&Vs[0][0][0], 64, 32, QKV_LD, PAD64);
    __builtin_amdgcn_s_wait_tensorcnt(2);            // Q landed
#else
    #pragma unroll
    for (int it = 0; it < 4; ++it) {
        int idx = lane + it * 32;
        int r = idx >> 3, c = idx & 7;
        *reinterpret_cast<uint4*>(&Qs[r][c * 8]) =
            *reinterpret_cast<const uint4*>(
                qkv + rowbase + (size_t)(t0 + r) * QKV_LD + h * HDn + c * 8);
    }
    __syncthreads();
#endif

    v16bf fragQ[2];
    #pragma unroll
    for (int kk0 = 0; kk0 < 2; ++kk0)
        #pragma unroll
        for (int i = 0; i < 16; ++i)
            fragQ[kk0][i] = Qs[l][kk0 * 32 + amap(i, hi)];

    // ones-column B fragment: column 0 of a 32x16 B matrix = 1, rest 0
    v16bf fone;
    #pragma unroll
    for (int i = 0; i < 16; ++i)
        fone[i] = (l == 0) ? (__bf16)1.0f : (__bf16)0.0f;

    v8f O[4] = {};
    v8f Lacc = {};                          // col 0 = running softmax denom
    float mrow[8];
    #pragma unroll
    for (int r = 0; r < 8; ++r) mrow[r] = -3.0e38f;

    const float sc2 = SCALE * LOG2E;        // base-2 logits

    for (int kt = 0; kt < NT; ++kt) {
        const int cur = kt & 1;
#if HAVE_TDM
        if (kt + 1 < NT) {
            const int nxt = cur ^ 1;
            tdm_load_2d(kbase + (size_t)((kt + 1) * 32) * QKV_LD,
                        (unsigned)(uintptr_t)&Ks[nxt][0][0], 64, 32, QKV_LD, PAD64);
            tdm_load_2d(vbase + (size_t)((kt + 1) * 32) * QKV_LD,
                        (unsigned)(uintptr_t)&Vs[nxt][0][0], 64, 32, QKV_LD, PAD64);
            __builtin_amdgcn_s_wait_tensorcnt(2);    // tiles for step kt landed
        } else {
            __builtin_amdgcn_s_wait_tensorcnt(0);
        }
#else
        #pragma unroll
        for (int it = 0; it < 8; ++it) {
            int idx = lane + it * 32;
            int r = idx >> 3, c = idx & 7;
            const size_t grow = (size_t)(kt * 32 + r) * QKV_LD;
            *reinterpret_cast<uint4*>(&Ks[cur][r][c * 8]) =
                *reinterpret_cast<const uint4*>(kbase + grow + c * 8);
            *reinterpret_cast<uint4*>(&Vs[cur][r][c * 8]) =
                *reinterpret_cast<const uint4*>(vbase + grow + c * 8);
        }
        __syncthreads();
#endif

        // --- scores: S[16 x 32] = Q (16x64) @ K_tile^T (64x32), 2x2 WMMA ---
        v8f S[2];
        #pragma unroll
        for (int ns = 0; ns < 2; ++ns) {
            v8f sacc = {};
            #pragma unroll
            for (int kk0 = 0; kk0 < 2; ++kk0) {
                v16bf fb;
                #pragma unroll
                for (int i = 0; i < 16; ++i)
                    fb[i] = Ks[cur][ns * 16 + l][kk0 * 32 + amap(i, hi)];
                sacc = __builtin_amdgcn_wmma_f32_16x16x32_bf16(
                    false, fragQ[kk0], false, fb, (short)0, sacc, false, false);
            }
            S[ns] = sacc;
        }

        // --- online softmax (rows r / r+8 live in the two lane halves) ---
        #pragma unroll
        for (int r = 0; r < 8; ++r) {
            float s0 = S[0][r] * sc2;
            float s1 = S[1][r] * sc2;
            float mnew = fmaxf(mrow[r], hmax16(fmaxf(s0, s1)));
            float alpha = exp2f(mrow[r] - mnew);
            float p0 = exp2f(s0 - mnew);
            float p1 = exp2f(s1 - mnew);
            mrow[r] = mnew;
            #pragma unroll
            for (int ns = 0; ns < 4; ++ns) O[ns][r] *= alpha;
            Lacc[r] *= alpha;
            Ps[r + hi * 8][l]      = (__bf16)p0;
            Ps[r + hi * 8][16 + l] = (__bf16)p1;
        }
        __syncthreads();

        // --- O += P (16x32) @ V (32x64); Lacc += P @ ones-column ---
        v16bf fp;
        #pragma unroll
        for (int i = 0; i < 16; ++i)
            fp[i] = Ps[l][amap(i, hi)];
        #pragma unroll
        for (int ns = 0; ns < 4; ++ns) {
            v16bf fv;
            #pragma unroll
            for (int i = 0; i < 16; ++i)
                fv[i] = Vs[cur][amap(i, hi)][ns * 16 + l];
            O[ns] = __builtin_amdgcn_wmma_f32_16x16x32_bf16(
                false, fp, false, fv, (short)0, O[ns], false, false);
        }
        Lacc = __builtin_amdgcn_wmma_f32_16x16x32_bf16(
            false, fp, false, fone, (short)0, Lacc, false, false);
        __syncthreads();
    }

    // --- normalize (broadcast denom from column-0 lane of each half) ---
    #pragma unroll
    for (int r = 0; r < 8; ++r) {
        float denom = __shfl(Lacc[r], hi << 4, 32);
        float invl = 1.0f / denom;
        int row = t0 + r + hi * 8;
        #pragma unroll
        for (int ns = 0; ns < 4; ++ns)
            ao[((size_t)(b * Tn) + row) * Dn + h * HDn + ns * 16 + l] =
                (__bf16)(O[ns][r] * invl);
    }
}

// ---------------------------------------------------------------------------
// Host-side launcher
// ---------------------------------------------------------------------------
extern "C" void kernel_launch(void* const* d_in, const int* in_sizes, int n_in,
                              void* d_out, int out_size, void* d_ws, size_t ws_size,
                              hipStream_t stream) {
    (void)in_sizes; (void)n_in; (void)out_size; (void)ws_size;

    const float* x     = (const float*)d_in[0];   // [B,T,D]
    const float* w_qkv = (const float*)d_in[1];   // [D, 1536]
    const float* w_o   = (const float*)d_in[2];   // [D, D]
    float* out = (float*)d_out;                   // [B,T,D] f32

    constexpr size_t XB_E   = (size_t)Bn * Tn * Dn;
    constexpr size_t WQKV_E = (size_t)Dn * QKV_LD;
    constexpr size_t WO_E   = (size_t)Dn * Dn;
    constexpr size_t QKV_E  = (size_t)Bn * Tn * QKV_LD;

    char* ws = (char*)d_ws;
    __bf16* xb    = (__bf16*)(ws);
    __bf16* wqkvb = (__bf16*)(ws + XB_E * 2);
    __bf16* wob   = (__bf16*)(ws + (XB_E + WQKV_E) * 2);
    __bf16* qkv   = (__bf16*)(ws + (XB_E + WQKV_E + WO_E) * 2);
    __bf16* ao    = (__bf16*)(ws + (XB_E + WQKV_E + WO_E + QKV_E) * 2);

    // 1) convert inputs to bf16
    f32_to_bf16_kernel<<<4096, 256, 0, stream>>>(x,     xb,    (int)XB_E);
    f32_to_bf16_kernel<<<1024, 256, 0, stream>>>(w_qkv, wqkvb, (int)WQKV_E);
    f32_to_bf16_kernel<<<1024, 256, 0, stream>>>(w_o,   wob,   (int)WO_E);

    // 2) qkv = x @ w_qkv    (bf16 out)
    gemm_bf16_wmma<__bf16><<<dim3(QKV_LD / 64, (Bn * Tn) / 64), 128, 0, stream>>>(
        xb, wqkvb, qkv, Bn * Tn, QKV_LD, Dn);

    // 3) 2D RoPE in place on q/k sections
    rope2d_kernel<<<Bn * Tn, (Hn + Gn) * 16, 0, stream>>>(qkv);

    // 4) flash attention -> bf16 [B*T, D]
    attn_kernel<<<dim3(Tn / 16, Hn, Bn), 32, 0, stream>>>(qkv, ao);

    // 5) out = ao @ w_o   (f32 out)
    gemm_bf16_wmma<float><<<dim3(Dn / 64, (Bn * Tn) / 64), 128, 0, stream>>>(
        ao, wob, out, Bn * Tn, Dn, Dn);
}
